// Layer_69252052680708
// MI455X (gfx1250) — compile-verified, matrix operand order checked by reading, not run on hardware
//
#include <hip/hip_runtime.h>
#include <hip/hip_bf16.h>
#include <math.h>

#define EPS_ 1e-4f
#define N_SAMP 32768
#define DIM 256
#define ZDIM 128
#define KOUT 256
#define KPAD 288          // 257 padded up to multiple of 32 (9 WMMA K-steps)
#define NEXP 16

typedef _Float16 v16h __attribute__((ext_vector_type(16)));
typedef _Float16 v8h  __attribute__((ext_vector_type(8)));
typedef float    v8f  __attribute__((ext_vector_type(8)));

// ---------------------------------------------------------------------------
// Load one 16x32 f16 fragment (A-layout; also valid for B as A-layout of B^T).
// Per ISA 7.12.2: lanes 0-15 hold row M=lane, K 0..7 in v[0..3], K 16..23 in
// v[4..7]; lanes 16-31 hold the same rows with K 8..15 / 24..31.
// Works for global (global_load_b128) and LDS (ds_load_b128) base pointers.
// ---------------------------------------------------------------------------
template <typename P>
__device__ __forceinline__ v16h load_frag(P base, int ld, int k0, int lane) {
  int m   = lane & 15;
  int sel = lane >> 4;           // 0 -> K {0..7,16..23}, 1 -> K {8..15,24..31}
  auto* p = base + (size_t)m * ld + k0 + sel * 8;
  union { v16h v; v8h h[2]; } u;
  u.h[0] = *reinterpret_cast<const v8h*>(p);
  u.h[1] = *reinterpret_cast<const v8h*>(p + 16);
  return u.v;
}

// ---------------------------------------------------------------------------
// Prep kernels: f32 -> f16 operand conversion (memory-bound, ~60 MB total)
// ---------------------------------------------------------------------------
__global__ void prep_logitx(const float* __restrict__ x, _Float16* __restrict__ lx) {
  size_t idx = (size_t)blockIdx.x * blockDim.x + threadIdx.x;  // over N*KPAD
  int n = (int)(idx / KPAD);
  int d = (int)(idx % KPAD);
  float v;
  if (d < DIM) {
    float xv = x[(size_t)n * DIM + d];
    v = logf(xv / (1.0f - xv));
  } else if (d == DIM) {
    v = logf((1.0f - EPS_) / EPS_);   // bias column: logit(1-eps)
  } else {
    v = 0.0f;                          // zero pad 257..287
  }
  lx[idx] = (_Float16)v;
}

__global__ void prep_w(const float* __restrict__ w, _Float16* __restrict__ w16) {
  size_t idx = (size_t)blockIdx.x * blockDim.x + threadIdx.x;  // over 4096*KPAD
  int r = (int)(idx / KPAD);
  int d = (int)(idx % KPAD);
  _Float16 v = (_Float16)0.0f;
  if (d <= DIM) v = (_Float16)w[(size_t)r * (DIM + 1) + d];
  w16[idx] = v;
}

__global__ void cvt_f16(const float* __restrict__ s, _Float16* __restrict__ d, int n) {
  int idx = blockIdx.x * blockDim.x + threadIdx.x;
  if (idx < n) d[idx] = (_Float16)s[idx];
}

// ---------------------------------------------------------------------------
// Gate kernel: gate = z16 @ gw16^T + b ; choice = sum_h (gate_h > 0) << h.
// Block = 8 waves sharing one k-column tile; the 4h x 16col x 128K B tile
// (16 KB) is staged in LDS once and reused by all 8 waves (8x less L2 read).
// Each wave: 16 samples x 16 cols, K=128 -> 16 v_wmma_f32_16x16x32_f16.
// ---------------------------------------------------------------------------
__global__ void gates_kernel(const _Float16* __restrict__ z16,
                             const _Float16* __restrict__ gw16,
                             const float*    __restrict__ gb,
                             unsigned char*  __restrict__ choices) {
  __shared__ _Float16 gsh[64 * ZDIM];            // (h*16+col) x K, 16 KB
  const int tid  = threadIdx.x;
  const int lane = tid & 31;
  const int wid  = tid >> 5;
  const int kT   = blockIdx.x & 15;
  const int nG   = blockIdx.x >> 4;              // 0..255
  const int row0 = (nG * 8 + wid) * 16;
  const int col0 = kT * 16;

  // cooperative fill: 64 rows x 128 halfs = 1024 v8h chunks, 4 per thread
#pragma unroll
  for (int c = tid; c < 64 * (ZDIM / 8); c += 256) {
    int r   = c >> 4;                            // 0..63 = h*16 + n
    int off = (c & 15) * 8;
    int g   = (r >> 4) * KOUT + col0 + (r & 15); // gw row
    *reinterpret_cast<v8h*>(&gsh[r * ZDIM + off]) =
        *reinterpret_cast<const v8h*>(gw16 + (size_t)g * ZDIM + off);
  }
  __syncthreads();

  const _Float16* A = z16 + (size_t)row0 * ZDIM;
  v16h a[4];
#pragma unroll
  for (int k = 0; k < 4; ++k) a[k] = load_frag(A, ZDIM, k * 32, lane);

  v8f acc[4];
#pragma unroll
  for (int h = 0; h < 4; ++h) {
    v8f c = {};
#pragma unroll
    for (int k = 0; k < 4; ++k) {
      v16h b = load_frag(&gsh[(h * 16) * ZDIM], ZDIM, k * 32, lane);
      c = __builtin_amdgcn_wmma_f32_16x16x32_f16(false, a[k], false, b,
                                                 (short)0, c, false, false);
    }
    acc[h] = c;
  }

  const int col = col0 + (lane & 15);
  const float b0 = gb[0 * KOUT + col];
  const float b1 = gb[1 * KOUT + col];
  const float b2 = gb[2 * KOUT + col];
  const float b3 = gb[3 * KOUT + col];
  const int rbase = row0 + ((lane >> 4) ? 8 : 0);
#pragma unroll
  for (int i = 0; i < 8; ++i) {
    int ch = ((acc[0][i] + b0) > 0.0f ? 1 : 0)
           | ((acc[1][i] + b1) > 0.0f ? 2 : 0)
           | ((acc[2][i] + b2) > 0.0f ? 4 : 0)
           | ((acc[3][i] + b3) > 0.0f ? 8 : 0);
    choices[(size_t)(rbase + i) * KOUT + col] = (unsigned char)ch;
  }
}

// ---------------------------------------------------------------------------
// Expert kernel: block = 8 waves (8 sample tiles) x one k-column tile.
// A (logit_x, 9 K-frags, 72 VGPRs) stays register-resident per wave.
// B (one expert's 16x288 f16 tile, 9.2 KB) is double-buffered in LDS via the
// CDNA5 async copy path (GLOBAL_LOAD_ASYNC_TO_LDS_B128, ASYNCcnt): the next
// expert's tile streams memory->LDS with no VGPR staging while the current
// expert's 9 WMMAs run. s_wait_asynccnt 0 drains the wave's async ops before
// the per-expert barrier (the barrier itself does not wait on ASYNCcnt).
// Per-element expert select + fused sigmoid/clip epilogue; the [N,4096]
// y tensor is never materialized.
// ---------------------------------------------------------------------------
__global__ void expert_kernel(const _Float16* __restrict__ lx,
                              const _Float16* __restrict__ w16,
                              const unsigned char* __restrict__ choices,
                              float* __restrict__ out) {
  __shared__ _Float16 bsh[2][16 * KPAD];         // 2 x 9216 B
  const int tid  = threadIdx.x;
  const int lane = tid & 31;
  const int wid  = tid >> 5;
  const int kT   = blockIdx.x & 15;
  const int nG   = blockIdx.x >> 4;
  const int row0 = (nG * 8 + wid) * 16;
  const int col0 = kT * 16;

  // Flat->LDS mapping truncates to addr[31:0] (allocation-relative), which is
  // exactly the LDS offset the async instruction takes in its VDST operand.
  const unsigned lds0 = (unsigned)(size_t)(&bsh[0][0]);
  const unsigned lds1 = (unsigned)(size_t)(&bsh[1][0]);

  // async fill: 16 rows x 288 halfs = 576 16-byte chunks, memory -> LDS
  auto fill_async = [&](int e, unsigned ldsbase) {
    const char* src = (const char*)(w16 + (size_t)(e * KOUT + col0) * KPAD);
    for (int c = tid; c < 16 * (KPAD / 8); c += 256) {
      int r   = c / (KPAD / 8);
      int off = (c % (KPAD / 8)) * 8;
      unsigned byteoff = (unsigned)((r * KPAD + off) * 2);
      const void* g = src + (size_t)byteoff;
      unsigned l = ldsbase + byteoff;
      asm volatile("global_load_async_to_lds_b128 %0, %1, off"
                   :: "v"(l), "v"(g) : "memory");
    }
  };

  const _Float16* A = lx + (size_t)row0 * KPAD;
  v16h a[9];
#pragma unroll
  for (int k = 0; k < 9; ++k) a[k] = load_frag(A, KPAD, k * 32, lane);

  const int col   = col0 + (lane & 15);
  const int rbase = row0 + ((lane >> 4) ? 8 : 0);
  int ch[8];
#pragma unroll
  for (int i = 0; i < 8; ++i)
    ch[i] = choices[(size_t)(rbase + i) * KOUT + col];

  fill_async(0, lds0);
  asm volatile("s_wait_asynccnt 0x0" ::: "memory");
  __syncthreads();

  v8f res = {};
#pragma unroll 1
  for (int e = 0; e < NEXP; ++e) {
    const int buf = e & 1;
    if (e + 1 < NEXP) fill_async(e + 1, buf ? lds0 : lds1);  // prefetch next
    v8f acc = {};
#pragma unroll
    for (int k = 0; k < 9; ++k) {
      v16h b = load_frag(&bsh[buf][0], KPAD, k * 32, lane);
      acc = __builtin_amdgcn_wmma_f32_16x16x32_f16(false, a[k], false, b,
                                                   (short)0, acc, false, false);
    }
#pragma unroll
    for (int i = 0; i < 8; ++i)
      res[i] = (ch[i] == e) ? acc[i] : res[i];
    asm volatile("s_wait_asynccnt 0x0" ::: "memory");  // my fill landed in LDS
    __syncthreads();                                   // everyone's fill done
  }

#pragma unroll
  for (int i = 0; i < 8; ++i) {
    float p = 1.0f / (1.0f + __expf(-res[i]));
    p = fminf(fmaxf(p, EPS_), 1.0f - EPS_);
    out[(size_t)(rbase + i) * KOUT + col] = p;
  }
}

// ---------------------------------------------------------------------------
// Launcher. Workspace layout (all offsets 256B aligned, total ~38.3 MB):
//   lx   : [N, 288] f16   @ 0          (18,874,368 B)
//   z16  : [N, 128] f16   @ 18,874,368 ( 8,388,608 B)
//   gw16 : [1024,128] f16 @ 27,262,976 (   262,144 B)
//   w16  : [4096,288] f16 @ 27,525,120 ( 2,359,296 B)
//   chc  : [N, 256] u8    @ 29,884,416 ( 8,388,608 B)
// ---------------------------------------------------------------------------
extern "C" void kernel_launch(void* const* d_in, const int* in_sizes, int n_in,
                              void* d_out, int out_size, void* d_ws, size_t ws_size,
                              hipStream_t stream) {
  const float* x  = (const float*)d_in[0];
  const float* z  = (const float*)d_in[1];
  const float* gw = (const float*)d_in[2];
  const float* gb = (const float*)d_in[3];
  const float* w  = (const float*)d_in[4];
  float* out = (float*)d_out;

  char* ws = (char*)d_ws;
  _Float16* lx   = (_Float16*)(ws);
  _Float16* z16  = (_Float16*)(ws + 18874368);
  _Float16* gw16 = (_Float16*)(ws + 27262976);
  _Float16* w16  = (_Float16*)(ws + 27525120);
  unsigned char* chc = (unsigned char*)(ws + 29884416);

  // operand conversion (memory-bound)
  prep_logitx<<<(N_SAMP * KPAD) / 256, 256, 0, stream>>>(x, lx);
  cvt_f16<<<(N_SAMP * ZDIM) / 256, 256, 0, stream>>>(z, z16, N_SAMP * ZDIM);
  cvt_f16<<<(4 * KOUT * ZDIM) / 256, 256, 0, stream>>>(gw, gw16, 4 * KOUT * ZDIM);
  prep_w<<<(NEXP * KOUT * KPAD) / 256, 256, 0, stream>>>(w, w16);

  // gate GEMM + choice bits (WMMA f16, LDS-shared B)
  gates_kernel<<<4096, 256, 0, stream>>>(z16, gw16, gb, chc);

  // expert GEMM + gather-select + sigmoid (WMMA f16, async double-buffered B)
  expert_kernel<<<4096, 256, 0, stream>>>(lx, w16, chc, out);
}
